// DenseFastGAT_56959856279566
// MI455X (gfx1250) — matrix operand order repaired and checked by reading, not compiled, not run on hardware
//
#include <hip/hip_runtime.h>
#include <hip/hip_bf16.h>

#define B_ 2
#define N_ 4096
#define F_ 256
#define ALPHA 0.2f
#define LOG2E 1.4426950408889634f
#define SINK2 (-1.0e9f * LOG2E)   /* sink logit, pre-scaled to log2 domain */

typedef __attribute__((ext_vector_type(16))) __bf16          v16bf;
typedef __attribute__((ext_vector_type(2)))  __bf16          v2bf;
typedef __attribute__((ext_vector_type(8)))  float           v8f;

__device__ __forceinline__ v16bf pack16(const float* f) {
#if __has_builtin(__builtin_amdgcn_cvt_pk_bf16_f32)
  union { v2bf h[8]; v16bf v; } u;
#pragma unroll
  for (int i = 0; i < 8; ++i)
    u.h[i] = __builtin_amdgcn_cvt_pk_bf16_f32(f[2 * i], f[2 * i + 1]);
  return u.v;
#else
  // plain fptrunc: lets the backend pick the native bf16 cvt if present
  v16bf v;
#pragma unroll
  for (int i = 0; i < 16; ++i) v[i] = (__bf16)f[i];
  return v;
#endif
}

__device__ __forceinline__ float fast_exp2(float x) {
#if __has_builtin(__builtin_amdgcn_exp2f)
  return __builtin_amdgcn_exp2f(x);        // v_exp_f32 (native exp2)
#else
  return exp2f(x);
#endif
}

// ---------------------------------------------------------------------------
// k0: compress fp32 adjacency (>0) into bitmask words. 1 thread = 1 u32 word
//     = 32 floats = one 128B cacheline -> perfectly streamed 128MB read.
// ---------------------------------------------------------------------------
__global__ __launch_bounds__(256) void k0_bits(const float* __restrict__ adj,
                                               unsigned* __restrict__ bits) {
  int w = blockIdx.x * blockDim.x + threadIdx.x;      // 0 .. B*N*N/32-1
  const float* p = adj + (size_t)w * 32;
  unsigned m = 0;
#pragma unroll
  for (int q = 0; q < 8; ++q) {
    float4 f = *(const float4*)(p + q * 4);
    m |= (f.x > 0.f ? 1u : 0u) << (q * 4 + 0);
    m |= (f.y > 0.f ? 1u : 0u) << (q * 4 + 1);
    m |= (f.z > 0.f ? 1u : 0u) << (q * 4 + 2);
    m |= (f.w > 0.f ? 1u : 0u) << (q * 4 + 3);
  }
  bits[w] = m;
}

// ---------------------------------------------------------------------------
// k1: z = x @ W^T + bW, bf16 WMMA with f32 accumulate.
//     One wave per 16x16 output tile; K loop of 8 x (16x16x32) WMMAs.
//     A (16-bit, 16x32): lane<16 row=lane, K={0..7,16..23}; lane>=16 same row,
//     K={8..15,24..31}. B (32x16): col=lane%16, same K sets per lane half as A.
// ---------------------------------------------------------------------------
__global__ __launch_bounds__(256) void k1_zgemm(const float* __restrict__ x,
                                                const float* __restrict__ W,
                                                const float* __restrict__ bW,
                                                float* __restrict__ z) {
  int wid  = (blockIdx.x * blockDim.x + threadIdx.x) >> 5;   // 0..8191
  int lane = threadIdx.x & 31;
  int ot = wid & 15, nt = (wid >> 4) & 255, b = wid >> 12;
  int half = lane >> 4, lo = lane & 15;
  int n = nt * 16 + lo;
  int o = ot * 16 + lo;
  const float* xrow = x + ((size_t)(b * N_ + n)) * F_;
  const float* wrow = W + (size_t)o * F_;
  v8f acc = {};
#pragma unroll
  for (int k0 = 0; k0 < F_; k0 += 32) {
    float a[16], bb[16];
    float4 a0 = *(const float4*)(xrow + k0 + half * 8);
    float4 a1 = *(const float4*)(xrow + k0 + half * 8 + 4);
    float4 a2 = *(const float4*)(xrow + k0 + 16 + half * 8);
    float4 a3 = *(const float4*)(xrow + k0 + 16 + half * 8 + 4);
    a[0]=a0.x; a[1]=a0.y; a[2]=a0.z; a[3]=a0.w;
    a[4]=a1.x; a[5]=a1.y; a[6]=a1.z; a[7]=a1.w;
    a[8]=a2.x; a[9]=a2.y; a[10]=a2.z; a[11]=a2.w;
    a[12]=a3.x; a[13]=a3.y; a[14]=a3.z; a[15]=a3.w;
    float4 b0 = *(const float4*)(wrow + k0 + half * 8);
    float4 b1 = *(const float4*)(wrow + k0 + half * 8 + 4);
    float4 b2 = *(const float4*)(wrow + k0 + 16 + half * 8);
    float4 b3 = *(const float4*)(wrow + k0 + 16 + half * 8 + 4);
    bb[0]=b0.x; bb[1]=b0.y; bb[2]=b0.z; bb[3]=b0.w;
    bb[4]=b1.x; bb[5]=b1.y; bb[6]=b1.z; bb[7]=b1.w;
    bb[8]=b2.x; bb[9]=b2.y; bb[10]=b2.z; bb[11]=b2.w;
    bb[12]=b3.x; bb[13]=b3.y; bb[14]=b3.z; bb[15]=b3.w;
    v16bf Av = pack16(a), Bv = pack16(bb);
    acc = __builtin_amdgcn_wmma_f32_16x16x32_bf16(false, Av, false, Bv,
                                                  (short)0, acc, false, false);
  }
  float bias = bW[o];
#pragma unroll
  for (int r = 0; r < 8; ++r) {
    int M = half * 8 + r;                // C layout: lanes 16-31 hold M+8
    z[((size_t)(b * N_ + nt * 16 + M)) * F_ + o] = acc[r] + bias;
  }
}

// ---------------------------------------------------------------------------
// k2: ai/aj row dot products. Results are PRE-SCALED by log2(e) so k4 can use
//     native exp2 (v_exp_f32) with no per-element multiply. Scaling by a
//     positive constant commutes with max and LeakyReLU.
// ---------------------------------------------------------------------------
__global__ __launch_bounds__(256) void k2_aiaj(const float* __restrict__ z,
                                               const float* __restrict__ wai,
                                               const float* __restrict__ bai,
                                               const float* __restrict__ waj,
                                               const float* __restrict__ baj,
                                               float* __restrict__ ai,
                                               float* __restrict__ aj) {
  int row  = (blockIdx.x * blockDim.x + threadIdx.x) >> 5;   // 0..8191
  int lane = threadIdx.x & 31;
  const float* zr = z + (size_t)row * F_;
  float si = 0.f, sj = 0.f;
#pragma unroll
  for (int t = 0; t < 8; ++t) {
    float v = zr[lane + t * 32];
    si += v * wai[lane + t * 32];
    sj += v * waj[lane + t * 32];
  }
#pragma unroll
  for (int off = 16; off > 0; off >>= 1) {
    si += __shfl_xor(si, off, 32);
    sj += __shfl_xor(sj, off, 32);
  }
  if (lane == 0) {
    ai[row] = (si + bai[0]) * LOG2E;
    aj[row] = (sj + baj[0]) * LOG2E;
  }
}

// ---------------------------------------------------------------------------
// k3: re-swizzle z (fp32) into B-operand-native bf16 layout:
//     zs[((b*128+jt)*16+ot)*32+lane][0..15] = z[b, jt*32 + kb + v + (v<8 ? 0 : 8), ot*16+lo]
//     One 32-byte vector store per thread.
// ---------------------------------------------------------------------------
__global__ __launch_bounds__(256) void k3_swz(const float* __restrict__ z,
                                              unsigned short* __restrict__ zs) {
  int t = blockIdx.x * blockDim.x + threadIdx.x;     // 0..131071
  int lane = t & 31, ot = (t >> 5) & 15, jt = (t >> 9) & 127, b = t >> 16;
  int kb = (lane >> 4) * 8, col = (lane & 15) + ot * 16;
  const float* src = z + ((size_t)(b * N_) + jt * 32 + kb) * F_ + col;
  float vals[16];
#pragma unroll
  for (int v = 0; v < 16; ++v) vals[v] = src[(size_t)(v < 8 ? v : v + 8) * F_];
  v16bf pk = pack16(vals);
  *(v16bf*)(zs + ((size_t)((b * 128 + jt) * 16 + ot) * 32 + lane) * 16) = pk;
}

// ---------------------------------------------------------------------------
// k4: fused masked softmax + P@Z. One wave = 16 rows x 128 cols.
//     Pass1: row max over edges (bitmask ctz walk; leaky applied once).
//     Pass2: per 32-j chunk build P tile in A-operand layout, then 9 WMMAs:
//       8 against z-tiles (numerator) + 1 against an all-ones tile, whose
//       C tile accumulates exact per-row sums of the quantized P -> softmax
//       denominator lands in the same lane/VGPR the scaling needs (no shfl).
//     All logits live in the log2 domain (ai/aj pre-scaled).
// ---------------------------------------------------------------------------
__global__ __launch_bounds__(128) void k4_attn(const unsigned* __restrict__ bits,
                                               const float* __restrict__ ai_g,
                                               const float* __restrict__ aj_g,
                                               const unsigned short* __restrict__ zs,
                                               float* __restrict__ out) {
  int wid  = (blockIdx.x * blockDim.x + threadIdx.x) >> 5;   // 0..1023
  int lane = threadIdx.x & 31;
  int cw = wid & 1, rt = (wid >> 1) & 255, b = wid >> 9;
  int half = lane >> 4, lo = lane & 15;
  int irow = rt * 16 + lo;
  const unsigned* brow = bits + ((size_t)(b * N_) + irow) * (N_ / 32);
  const float* ajb = aj_g + (size_t)b * N_;
  float aival = ai_g[(size_t)b * N_ + irow];

  // ---- pass 1: max over edges of (ai+aj); leaky is monotone ----
  float vmax = -3.0e38f;
  for (int w = half * 64; w < half * 64 + 64; ++w) {
    unsigned u = brow[w];
    while (u) {
      int k = __builtin_ctz(u);
      u &= u - 1;
      vmax = fmaxf(vmax, aival + ajb[w * 32 + k]);
    }
  }
  vmax = fmaxf(vmax, __shfl_xor(vmax, 16, 32));
  float em = fmaxf(vmax, 0.f) + ALPHA * fminf(vmax, 0.f);   // leaky, branch-free
  float m = fmaxf(em, SINK2);                    // sink participates in max

  // ---- pass 2: P tile -> 9 WMMAs per 32-j chunk ----
  v8f acc[8] = {};
  v8f accd = {};                                 // denominator via P @ ones
  v16bf ones;
#pragma unroll
  for (int i = 0; i < 16; ++i) ones[i] = (__bf16)1.0f;

  const unsigned short* zsb = zs + (size_t)b * 128 * 16 * 32 * 16;
  for (int jc = 0; jc < 128; ++jc) {
    unsigned u = brow[jc];
    int j0 = jc * 32;
    // lane's 16 aj values: two contiguous 8-float groups, vectorized
    float4 aj0 = *(const float4*)(ajb + j0 + half * 8);
    float4 aj1 = *(const float4*)(ajb + j0 + half * 8 + 4);
    float4 aj2 = *(const float4*)(ajb + j0 + 16 + half * 8);
    float4 aj3 = *(const float4*)(ajb + j0 + 16 + half * 8 + 4);
    float ajv[16] = {aj0.x, aj0.y, aj0.z, aj0.w, aj1.x, aj1.y, aj1.z, aj1.w,
                     aj2.x, aj2.y, aj2.z, aj2.w, aj3.x, aj3.y, aj3.z, aj3.w};
    float p[16];
#pragma unroll
    for (int s = 0; s < 16; ++s) {
      int kk = (s < 8) ? (half * 8 + s) : (16 + half * 8 + (s - 8));
      float t = aival + ajv[s];
      float e = fmaxf(t, 0.f) + ALPHA * fminf(t, 0.f);
      float pe = fast_exp2(e - m);                 // e<=m, no overflow
      p[s] = ((u >> kk) & 1u) ? pe : 0.f;          // non-edge -> exactly 0
    }
    v16bf Av = pack16(p);
    const v16bf* Bp = (const v16bf*)(zsb + ((size_t)jc * 16) * 32 * 16);
    // prefetch next chunk's B tile and aj block into cache
    __builtin_prefetch((const void*)(Bp + 16 * 32 + lane), 0, 0);
    __builtin_prefetch((const void*)(ajb + j0 + 32 + half * 8), 0, 0);
    accd = __builtin_amdgcn_wmma_f32_16x16x32_bf16(false, Av, false, ones,
                                                   (short)0, accd, false, false);
#pragma unroll
    for (int ct = 0; ct < 8; ++ct) {
      int otg = cw * 8 + ct;
      v16bf Bv = Bp[(size_t)otg * 32 + lane];
      acc[ct] = __builtin_amdgcn_wmma_f32_16x16x32_bf16(false, Av, false, Bv,
                                                        (short)0, acc[ct],
                                                        false, false);
    }
  }
  // denominator: accd[r] holds the full row sum for row M=half*8+r already in
  // this lane (C layout); only the sink term needs the row max via shuffle.
  float inv[8];
#pragma unroll
  for (int r = 0; r < 8; ++r) {
    float mr = __shfl(m, half * 8 + r, 32);
    inv[r] = 1.0f / (accd[r] + fast_exp2(SINK2 - mr));
  }
#pragma unroll
  for (int ct = 0; ct < 8; ++ct) {
    int col = cw * 128 + ct * 16 + lo;
#pragma unroll
    for (int r = 0; r < 8; ++r) {
      int M = half * 8 + r;
      out[((size_t)(b * N_) + rt * 16 + M) * F_ + col] = acc[ct][r] * inv[r];
    }
  }
}

// ---------------------------------------------------------------------------
extern "C" void kernel_launch(void* const* d_in, const int* in_sizes, int n_in,
                              void* d_out, int out_size, void* d_ws,
                              size_t ws_size, hipStream_t stream) {
  (void)in_sizes; (void)n_in; (void)out_size; (void)ws_size;
  const float* x    = (const float*)d_in[0];
  const float* adjs = (const float*)d_in[1];
  const float* W    = (const float*)d_in[2];
  const float* bW   = (const float*)d_in[3];
  const float* wai  = (const float*)d_in[4];
  const float* bai  = (const float*)d_in[5];
  const float* waj  = (const float*)d_in[6];
  const float* baj  = (const float*)d_in[7];
  float* out = (float*)d_out;

  char* ws = (char*)d_ws;
  float*          z    = (float*)ws;                                // 8 MB
  unsigned short* zsw  = (unsigned short*)(ws + ((size_t)8 << 20)); // 4 MB
  unsigned*       bits = (unsigned*)(ws + ((size_t)12 << 20));      // 4 MB
  float*          ai   = (float*)(ws + ((size_t)16 << 20));         // 32 KB
  float*          aj   = ai + B_ * N_;                              // 32 KB

  k0_bits <<<4096, 256, 0, stream>>>(adjs, bits);
  k1_zgemm<<<1024, 256, 0, stream>>>(x, W, bW, z);
  k2_aiaj <<<1024, 256, 0, stream>>>(z, wai, bai, waj, baj, ai, aj);
  k3_swz  <<<512,  256, 0, stream>>>(z, zsw);
  k4_attn <<<256,  128, 0, stream>>>(bits, ai, aj, zsw, out);
}
